// Conv1d_76158360093303
// MI455X (gfx1250) — matrix operand order repaired and checked by reading, not compiled, not run on hardware
//
#include <hip/hip_runtime.h>

#define IN_CH   256
#define OUT_CH  512
#define LEN     4096
#define KW      9
#define PADW    4
#define BATCH   32
#define KTOT    (IN_CH * KW)        // 2304
#define NCHUNK  (KTOT / 32)         // 72 K-chunks of 32
#define NTILE   128                 // output positions per block
#define MTILE   128                 // output channels per block
#define XS_W    (NTILE + 2*PADW)    // 136 staged positions (halo)
#define XSTRIDE 264                 // padded LDS row stride in halfs (132 dwords, %64 = 4)

typedef __bf16         v16bf __attribute__((ext_vector_type(16)));
typedef float          v8f   __attribute__((ext_vector_type(8)));
typedef unsigned short us8   __attribute__((ext_vector_type(8)));

// f32 -> bf16 bits, round-to-nearest-even
static __device__ __forceinline__ unsigned short f2us(float f) {
    unsigned u = __builtin_bit_cast(unsigned, f);
    u += 0x7FFFu + ((u >> 16) & 1u);
    return (unsigned short)(u >> 16);
}

static __device__ __forceinline__ v16bf cat16(us8 lo, us8 hi) {
    return __builtin_bit_cast(
        v16bf, __builtin_shufflevector(lo, hi, 0, 1, 2, 3, 4, 5, 6, 7,
                                               8, 9, 10, 11, 12, 13, 14, 15));
}

// ---------------------------------------------------------------------------
// Pre-pack weights into exact bf16 A-fragment order, K reordered as k=t*256+c.
// Layout: wpk[ ((m*NCHUNK + kk)*32 + lane)*16 + e ]  (m = 16-row o-tile index)
// Element e: v=e>>1, p=e&1 -> Klocal = (v<4 ? 2v : 8+2v) + p + ((lane>=16)?8:0)
// t = kk/8 (constant per chunk), c = (kk%8)*32 + Klocal, o = m*16 + (lane&15)
// Size: 32*72*32*16 halfs = 2.25 MB in d_ws.
// ---------------------------------------------------------------------------
__global__ __launch_bounds__(256)
void pack_weights(const float* __restrict__ w, unsigned short* __restrict__ wpk) {
    int gid  = blockIdx.x * 256 + threadIdx.x;
    int e    = gid & 15;
    int l    = (gid >> 4) & 31;
    int rest = gid >> 9;
    int kk   = rest % NCHUNK;
    int m    = rest / NCHUNK;

    int v  = e >> 1, p = e & 1;
    int kl = ((v < 4) ? (2 * v) : (8 + 2 * v)) + p + ((l >> 4) << 3);
    int t  = kk >> 3;
    int c  = ((kk & 7) << 5) + kl;
    int o  = m * 16 + (l & 15);

    wpk[gid] = f2us(w[(size_t)o * KTOT + c * KW + t]);
}

// ---------------------------------------------------------------------------
// Implicit-GEMM conv1d: M=out_ch, N=(b,i), K=t*256+c, bf16 WMMA / f32 acc.
// Block = 256 threads (8 waves) computes 128(o) x 128(i) for one batch.
// Wave grid 4(M) x 2(N): wave owns m-tiles {wm, wm+4} and i-subtiles
// wn*64 .. wn*64+63 -> per chunk: 2 A frags (4 glb b128), 4 B frags
// (8 ds b128), 8 WMMA; each B fragment feeds 2 WMMAs.
// ---------------------------------------------------------------------------
__global__ __launch_bounds__(256)
void conv1d_wmma_bf16(const float* __restrict__ x,
                      const unsigned short* __restrict__ wpk,
                      const float* __restrict__ bias,
                      float* __restrict__ out) {
    // transposed bf16 x tile: xs[j][c], j in [0,136), padded row stride 264
    __shared__ unsigned short xs[XS_W * XSTRIDE];

    const int tid  = threadIdx.x;
    const int wv   = tid >> 5;
    const int lane = tid & 31;
    const int wm   = wv & 3;     // M-pair selector (m-tiles wm and wm+4)
    const int wn   = wv >> 2;    // N-half selector (i-subtiles 4wn..4wn+3)
    const int i0 = blockIdx.x * NTILE;
    const int o0 = blockIdx.y * MTILE;
    const int b  = blockIdx.z;

    // ---- stage x tile transposed into LDS (float4 reads, bf16 u16 writes) ----
    const float* xb = x + (size_t)b * IN_CH * LEN;
    for (int idx = tid; idx < IN_CH * (XS_W / 4); idx += 256) {
        int c   = idx / (XS_W / 4);
        int jv  = idx - c * (XS_W / 4);
        int pos = i0 - PADW + 4 * jv;          // 16B-aligned; OOB only whole-vec
        float4 v = make_float4(0.f, 0.f, 0.f, 0.f);
        if (pos >= 0 && pos < LEN)
            v = *(const float4*)(xb + (size_t)c * LEN + pos);
        int j = 4 * jv;
        xs[(j + 0) * XSTRIDE + c] = f2us(v.x);
        xs[(j + 1) * XSTRIDE + c] = f2us(v.y);
        xs[(j + 2) * XSTRIDE + c] = f2us(v.z);
        xs[(j + 3) * XSTRIDE + c] = f2us(v.w);
    }
    __syncthreads();

    const int mlane = lane & 15;
    const int hi    = lane >> 4;

    v8f acc0[4] = {};   // m-tile wm
    v8f acc1[4] = {};   // m-tile wm+4

    // A fragments: pre-packed, per-lane contiguous 32B, wave reads 1KB/chunk/tile
    const int m0 = blockIdx.y * 8 + wm;
    const unsigned short* aptr0 =
        wpk + (((size_t)m0 * NCHUNK) * 32 + (size_t)lane) * 16;
    const unsigned short* aptr1 = aptr0 + (size_t)4 * NCHUNK * 512;

    for (int kk = 0; kk < NCHUNK; ++kk) {
        __builtin_prefetch(aptr0 + 512, 0, 3);  // next chunk (L2-resident)
        __builtin_prefetch(aptr1 + 512, 0, 3);
        v16bf a0 = cat16(*(const us8*)(aptr0), *(const us8*)(aptr0 + 8));
        v16bf a1 = cat16(*(const us8*)(aptr1), *(const us8*)(aptr1 + 8));
        aptr0 += 512;
        aptr1 += 512;

        const int t  = kk >> 3;                // fixed per chunk
        const int c0 = (kk & 7) << 5;          // 32 consecutive channels
        const unsigned short* bb =
            &xs[(mlane + t + wn * 64) * XSTRIDE + c0 + (hi << 3)];

        #pragma unroll
        for (int u = 0; u < 4; ++u) {
            us8 b0 = *(const us8*)(bb + u * 16 * XSTRIDE);
            us8 b1 = *(const us8*)(bb + u * 16 * XSTRIDE + 16);
            v16bf bf = cat16(b0, b1);
            acc0[u] = __builtin_amdgcn_wmma_f32_16x16x32_bf16(
                false, a0, false, bf, (short)0, acc0[u], false, false);
            acc1[u] = __builtin_amdgcn_wmma_f32_16x16x32_bf16(
                false, a1, false, bf, (short)0, acc1[u], false, false);
        }
    }

    // ---- epilogue: C/D layout -> out[b, o, i] + bias ----
    const int orow0 = o0 + wm * 16 + hi * 8;   // m-tile wm
    const int orow1 = orow0 + 64;              // m-tile wm+4
    float bs0[8], bs1[8];
    #pragma unroll
    for (int r = 0; r < 8; ++r) { bs0[r] = bias[orow0 + r]; bs1[r] = bias[orow1 + r]; }

    const int icol = i0 + wn * 64 + mlane;
    float* ob0 = out + ((size_t)b * OUT_CH + orow0) * LEN + icol;
    float* ob1 = out + ((size_t)b * OUT_CH + orow1) * LEN + icol;
    #pragma unroll
    for (int r = 0; r < 8; ++r) {
        #pragma unroll
        for (int u = 0; u < 4; ++u) {
            ob0[(size_t)r * LEN + u * 16] = acc0[u][r] + bs0[r];
            ob1[(size_t)r * LEN + u * 16] = acc1[u][r] + bs1[r];
        }
    }
}

extern "C" void kernel_launch(void* const* d_in, const int* in_sizes, int n_in,
                              void* d_out, int out_size, void* d_ws, size_t ws_size,
                              hipStream_t stream) {
    const float* x    = (const float*)d_in[0];
    const float* wgt  = (const float*)d_in[1];
    const float* bias = (const float*)d_in[2];
    float* out        = (float*)d_out;
    unsigned short* wpk = (unsigned short*)d_ws;  // 2.25 MB packed bf16 weights

    // 1) pack weights into WMMA A-fragment layout (runs every call; stream-ordered)
    const int pack_elems = (OUT_CH / 16) * NCHUNK * 32 * 16;  // 1,179,648
    pack_weights<<<pack_elems / 256, 256, 0, stream>>>(wgt, wpk);

    // 2) main conv kernel
    dim3 grid(LEN / NTILE, OUT_CH / MTILE, BATCH);  // 32 x 4 x 32
    conv1d_wmma_bf16<<<grid, 256, 0, stream>>>(x, wpk, bias, out);
}
